// Adaptive_Tied_Input_Softmax_68607807587023
// MI455X (gfx1250) — matrix operand order repaired and checked by reading, not compiled, not run on hardware
//
#include <hip/hip_runtime.h>
#include <hip/hip_bf16.h>

typedef __attribute__((ext_vector_type(16))) _Float16 v16h;
typedef __attribute__((ext_vector_type(8)))  _Float16 v8h;
typedef __attribute__((ext_vector_type(8)))  float    v8f;
typedef __attribute__((ext_vector_type(4)))  float    v4f;

#define DIMSZ 1024
#define NTOK  1024      // B*S = 2*512
#define VTOT  100000

// ---------------------------------------------------------------------------
// f32 -> f16 elementwise convert
// ---------------------------------------------------------------------------
__global__ __launch_bounds__(256) void cvt_f32_f16(const float* __restrict__ in,
                                                   _Float16* __restrict__ out, int n) {
    int i = blockIdx.x * 256 + threadIdx.x;
    if (i < n) out[i] = (_Float16)in[i];
}

// ---------------------------------------------------------------------------
// embeds[3] [rows x 16] f32 -> [rows x 32] f16, zero-padded in K
// ---------------------------------------------------------------------------
__global__ __launch_bounds__(256) void pad_emb3(const float* __restrict__ e3,
                                                _Float16* __restrict__ out, int rows) {
    int i = blockIdx.x * 256 + threadIdx.x;
    if (i < rows * 32) {
        int r = i >> 5, c = i & 31;
        out[i] = (c < 16) ? (_Float16)e3[r * 16 + c] : (_Float16)0.f;
    }
}

// ---------------------------------------------------------------------------
// t_l = hidden @ projs[l]  (l=1..3, f16 out, l=3 padded to K=32)
// l_tail = hidden @ cls_w^T + cls_b  (f32)
// 8 tokens per block staged in LDS so each proj column load feeds 8 FMAs.
// ---------------------------------------------------------------------------
#define TPB 8
__global__ __launch_bounds__(256) void compute_t(
    const float* __restrict__ hidden, const float* __restrict__ p1,
    const float* __restrict__ p2, const float* __restrict__ p3,
    const float* __restrict__ clsw, const float* __restrict__ clsb,
    _Float16* __restrict__ t1, _Float16* __restrict__ t2,
    _Float16* __restrict__ t3, float* __restrict__ tail) {
    __shared__ float h[TPB][DIMSZ];
    const int tid = threadIdx.x;
    const int m0 = blockIdx.x * TPB;
    for (int i = tid; i < TPB * DIMSZ; i += 256)
        h[i >> 10][i & 1023] = hidden[(size_t)m0 * DIMSZ + i];
    __syncthreads();

    {   // t1: vd = 256
        float acc[TPB];
#pragma unroll
        for (int t = 0; t < TPB; ++t) acc[t] = 0.f;
        const int j = tid;
#pragma unroll 4
        for (int d = 0; d < DIMSZ; ++d) {
            float p = p1[d * 256 + j];
#pragma unroll
            for (int t = 0; t < TPB; ++t) acc[t] += h[t][d] * p;
        }
#pragma unroll
        for (int t = 0; t < TPB; ++t) t1[(size_t)(m0 + t) * 256 + j] = (_Float16)acc[t];
    }
    if (tid < 64) {  // t2: vd = 64
        float acc[TPB];
#pragma unroll
        for (int t = 0; t < TPB; ++t) acc[t] = 0.f;
#pragma unroll 4
        for (int d = 0; d < DIMSZ; ++d) {
            float p = p2[d * 64 + tid];
#pragma unroll
            for (int t = 0; t < TPB; ++t) acc[t] += h[t][d] * p;
        }
#pragma unroll
        for (int t = 0; t < TPB; ++t) t2[(size_t)(m0 + t) * 64 + tid] = (_Float16)acc[t];
    }
    if (tid < 32) {  // t3: vd = 16, padded to 32 with zeros
        float acc[TPB];
#pragma unroll
        for (int t = 0; t < TPB; ++t) acc[t] = 0.f;
        if (tid < 16) {
#pragma unroll 4
            for (int d = 0; d < DIMSZ; ++d) {
                float p = p3[d * 16 + tid];
#pragma unroll
                for (int t = 0; t < TPB; ++t) acc[t] += h[t][d] * p;
            }
        }
#pragma unroll
        for (int t = 0; t < TPB; ++t) t3[(size_t)(m0 + t) * 32 + tid] = (_Float16)acc[t];
    }
    if (tid < 3) {  // l_tail
        float acc[TPB];
#pragma unroll
        for (int t = 0; t < TPB; ++t) acc[t] = 0.f;
#pragma unroll 4
        for (int d = 0; d < DIMSZ; ++d) {
            float w = clsw[tid * DIMSZ + d];
#pragma unroll
            for (int t = 0; t < TPB; ++t) acc[t] += h[t][d] * w;
        }
        float b = clsb[tid];
#pragma unroll
        for (int t = 0; t < TPB; ++t) tail[(m0 + t) * 3 + tid] = acc[t] + b;
    }
}

// ---------------------------------------------------------------------------
// Embedding gather: out[tok, d] = sum_v E_l[local, v] * P_l[d, v]
// ---------------------------------------------------------------------------
__global__ __launch_bounds__(256) void embed_gather(
    const int* __restrict__ input,
    const float* __restrict__ e0, const float* __restrict__ e1,
    const float* __restrict__ e2, const float* __restrict__ e3,
    const float* __restrict__ p0, const float* __restrict__ p1,
    const float* __restrict__ p2, const float* __restrict__ p3,
    float* __restrict__ out) {
    __shared__ float ev[DIMSZ];
    const int tok = blockIdx.x;
    int id = input[tok];
    const float* E; const float* P; int vd;
    if (id < 20000)      { E = e0; P = p0; vd = 1024; }
    else if (id < 40000) { E = e1; P = p1; vd = 256;  id -= 20000; }
    else if (id < 70000) { E = e2; P = p2; vd = 64;   id -= 40000; }
    else                 { E = e3; P = p3; vd = 16;   id -= 70000; }
    for (int v = threadIdx.x; v < vd; v += 256) ev[v] = E[(size_t)id * vd + v];
    __syncthreads();
#pragma unroll
    for (int rr = 0; rr < 4; ++rr) {
        int d = threadIdx.x + 256 * rr;
        const float* pr = P + (size_t)d * vd;
        float acc = 0.f;
        for (int v = 0; v < vd; v += 4) {
            v4f pv = *(const v4f*)(pr + v);
            acc += ev[v] * pv[0] + ev[v + 1] * pv[1] + ev[v + 2] * pv[2] + ev[v + 3] * pv[3];
        }
        out[(size_t)tok * DIMSZ + d] = acc;
    }
}

// ---------------------------------------------------------------------------
// Main logits GEMM: C[m,n] = sum_k A[m,k] * E[n,k]  (+ per-row tail bias)
// A: [1024 x K] f16 row-major.  E: [V x K] row-major, f32 (converted on load)
// or f16 (bucket 3).  Block 256 thr = 8 waves (2 M x 4 N), tile 128x256,
// wave tile 64x64 = 16 v_wmma_f32_16x16x32_f16 per K-step.
// Out-of-range B rows are CLAMPED for addressing (branchless K-loop);
// the garbage lanes are finite and dropped by the guarded store.
// ---------------------------------------------------------------------------
template <bool B_IS_F16>
__global__ __launch_bounds__(256) void gemm_logits(
    const _Float16* __restrict__ A, const _Float16* __restrict__ Bh,
    const float* __restrict__ Bf, float* __restrict__ out, int V, int K,
    const float* __restrict__ tail, int tailIdx) {
    const int tid  = threadIdx.x;
    const int lane = tid & 31;
    const int wave = tid >> 5;
    const int wm = wave & 1;   // 0..1
    const int wn = wave >> 1;  // 0..3
    const int r  = lane & 15;
    const int g  = lane >> 4;
    const int bm = blockIdx.y * 128 + wm * 64;
    const int bn = blockIdx.x * 256 + wn * 64;

    v8f acc[4][4];
#pragma unroll
    for (int mt = 0; mt < 4; ++mt)
#pragma unroll
        for (int nt = 0; nt < 4; ++nt)
#pragma unroll
            for (int e = 0; e < 8; ++e) acc[mt][nt][e] = 0.f;

    int arow[4], bcol[4], bload[4]; bool bvalid[4];
#pragma unroll
    for (int mt = 0; mt < 4; ++mt) arow[mt] = bm + mt * 16 + r;
#pragma unroll
    for (int nt = 0; nt < 4; ++nt) {
        bcol[nt]   = bn + nt * 16 + r;
        bvalid[nt] = bcol[nt] < V;
        bload[nt]  = bvalid[nt] ? bcol[nt] : (V - 1);  // clamped load row
    }

    for (int kb = 0; kb < K; kb += 32) {
        v16h a[4], b[4];
#pragma unroll
        for (int mt = 0; mt < 4; ++mt) {
            // A fragment: lane (m=r, g): K = [kb+16g, +16) contiguous
            const _Float16* p = A + (size_t)arow[mt] * K + kb;
            v8h lo = *(const v8h*)(p + 16 * g);
            v8h hi = *(const v8h*)(p + 16 * g + 8);
#pragma unroll
            for (int e = 0; e < 8; ++e) { a[mt][e] = lo[e]; a[mt][e + 8] = hi[e]; }
        }
#pragma unroll
        for (int nt = 0; nt < 4; ++nt) {
            // B fragment: lane (n=r, g) holds column n, K = [kb+16g, +16) contiguous
            if (B_IS_F16) {
                const _Float16* p = Bh + (size_t)bload[nt] * K + kb + 16 * g;
                v8h lo = *(const v8h*)p;
                v8h hi = *(const v8h*)(p + 8);
#pragma unroll
                for (int e = 0; e < 8; ++e) { b[nt][e] = lo[e]; b[nt][e + 8] = hi[e]; }
            } else {
                const float* p = Bf + (size_t)bload[nt] * K + kb + 16 * g;
                v4f f0 = *(const v4f*)(p);
                v4f f1 = *(const v4f*)(p + 4);
                v4f f2 = *(const v4f*)(p + 8);
                v4f f3 = *(const v4f*)(p + 12);
#pragma unroll
                for (int e = 0; e < 4; ++e) {
                    b[nt][e]      = (_Float16)f0[e];
                    b[nt][e + 4]  = (_Float16)f1[e];
                    b[nt][e + 8]  = (_Float16)f2[e];
                    b[nt][e + 12] = (_Float16)f3[e];
                }
            }
        }
#pragma unroll
        for (int mt = 0; mt < 4; ++mt)
#pragma unroll
            for (int nt = 0; nt < 4; ++nt)
                acc[mt][nt] = __builtin_amdgcn_wmma_f32_16x16x32_f16(
                    false, a[mt], false, b[nt], (short)0, acc[mt][nt], false, false);
    }

    // C/D layout: VGPR j, lane (r,g) -> m = j + 8g, n = r
#pragma unroll
    for (int mt = 0; mt < 4; ++mt) {
#pragma unroll
        for (int j = 0; j < 8; ++j) {
            int m = bm + mt * 16 + j + 8 * g;
            float bias = tail ? tail[m * 3 + tailIdx] : 0.f;
#pragma unroll
            for (int nt = 0; nt < 4; ++nt) {
                if (bvalid[nt])
                    out[(size_t)m * VTOT + bcol[nt]] = acc[mt][nt][j] + bias;
            }
        }
    }
}

// ---------------------------------------------------------------------------
extern "C" void kernel_launch(void* const* d_in, const int* in_sizes, int n_in,
                              void* d_out, int out_size, void* d_ws, size_t ws_size,
                              hipStream_t stream) {
    const int*   input  = (const int*)d_in[0];
    const float* hidden = (const float*)d_in[1];
    const float* e0 = (const float*)d_in[2];
    const float* e1 = (const float*)d_in[3];
    const float* e2 = (const float*)d_in[4];
    const float* e3 = (const float*)d_in[5];
    const float* p0 = (const float*)d_in[6];
    const float* p1 = (const float*)d_in[7];
    const float* p2 = (const float*)d_in[8];
    const float* p3 = (const float*)d_in[9];
    const float* clsw = (const float*)d_in[10];
    const float* clsb = (const float*)d_in[11];

    float* out_embed = (float*)d_out;
    float* logits    = out_embed + (size_t)NTOK * DIMSZ;

    // workspace carve-out (~4.8 MB)
    uintptr_t off = (uintptr_t)d_ws;
    auto carve = [&](size_t bytes) {
        void* p = (void*)off;
        off = (off + bytes + 255) & ~(uintptr_t)255;
        return p;
    };
    _Float16* hid_h  = (_Float16*)carve((size_t)NTOK * 1024 * 2);
    _Float16* t1_h   = (_Float16*)carve((size_t)NTOK * 256 * 2);
    _Float16* t2_h   = (_Float16*)carve((size_t)NTOK * 64 * 2);
    _Float16* t3_h   = (_Float16*)carve((size_t)NTOK * 32 * 2);  // K padded to 32
    _Float16* emb3_h = (_Float16*)carve((size_t)30000 * 32 * 2); // K padded to 32
    float*    tail   = (float*)carve((size_t)NTOK * 3 * 4);

    // 1) hidden -> f16
    cvt_f32_f16<<<(NTOK * 1024 + 255) / 256, 256, 0, stream>>>(hidden, hid_h, NTOK * 1024);
    // 2) embeds[3] -> f16 padded
    pad_emb3<<<(30000 * 32 + 255) / 256, 256, 0, stream>>>(e3, emb3_h, 30000);
    // 3) t_l and l_tail
    compute_t<<<NTOK / TPB, 256, 0, stream>>>(hidden, p1, p2, p3, clsw, clsb,
                                              t1_h, t2_h, t3_h, tail);
    // 4) embedding gather
    embed_gather<<<NTOK, 256, 0, stream>>>(input, e0, e1, e2, e3, p0, p1, p2, p3, out_embed);

    // 5) logits GEMMs per bucket (col offsets 0 / 20000 / 40000 / 70000)
    gemm_logits<false><<<dim3((20000 + 255) / 256, NTOK / 128), 256, 0, stream>>>(
        hid_h, nullptr, e0, logits + 0, 20000, 1024, nullptr, 0);
    gemm_logits<false><<<dim3((20000 + 255) / 256, NTOK / 128), 256, 0, stream>>>(
        t1_h, nullptr, e1, logits + 20000, 20000, 256, tail, 0);
    gemm_logits<false><<<dim3((30000 + 255) / 256, NTOK / 128), 256, 0, stream>>>(
        t2_h, nullptr, e2, logits + 40000, 30000, 64, tail, 1);
    gemm_logits<true><<<dim3((30000 + 255) / 256, NTOK / 128), 256, 0, stream>>>(
        t3_h, emb3_h, nullptr, logits + 70000, 30000, 32, tail, 2);
}
